// SSLModule_29025388987032
// MI455X (gfx1250) — matrix operand order repaired and checked by reading, not compiled, not run on hardware
//
#include <hip/hip_runtime.h>

#define B_ROWS 4096
#define D_IN   256
#define H_DIM  256
#define P_DIM  128
#define N_TOT  8192
#define INV_T  (1.0f / 0.07f)

typedef float    v2f  __attribute__((ext_vector_type(2)));
typedef float    v8f  __attribute__((ext_vector_type(8)));
typedef _Float16 v8h  __attribute__((ext_vector_type(8)));
typedef _Float16 v16h __attribute__((ext_vector_type(16)));

// Exact fp32 path: D = A(16x4 f32) * B(4x16 f32) + C  (projection GEMMs)
__device__ __forceinline__ v8f wmma_f32(v2f a, v2f b, v8f c) {
    return __builtin_amdgcn_wmma_f32_16x16x4_f32(
        false, a, false, b, (short)0, c, false, false);
}
// Throughput path: D = A(16x32 f16) * B(32x16 f16) + C(f32)  (similarity GEMM)
__device__ __forceinline__ v8f wmma_f16(v16h a, v16h b, v8f c) {
    return __builtin_amdgcn_wmma_f32_16x16x32_f16(
        false, a, false, b, (short)0, c, false, false);
}

// ---------------------------------------------------------------------------
// Kernel 1: fused projection MLP + L2 normalize (full fp32 WMMA precision),
// emits f16 features for the throughput GEMM. One wave per 16 rows.
// ---------------------------------------------------------------------------
__global__ __launch_bounds__(32) void proj_kernel(
    const float* __restrict__ x1, const float* __restrict__ x2,
    const float* __restrict__ W1, const float* __restrict__ b1,
    const float* __restrict__ W2, const float* __restrict__ b2,
    _Float16* __restrict__ F16)
{
    __shared__ float Hs[16 * H_DIM];            // 16 KB

    const int lane = threadIdx.x;               // 0..31 (wave32)
    const int lh   = lane & 15;
    const int koff = (lane >> 4) * 2;           // f32 A/B fragment K offset
    const int r0   = blockIdx.x * 16;

    const float* X   = (r0 < B_ROWS) ? x1 : x2;
    const int    xr0 = (r0 < B_ROWS) ? r0 : (r0 - B_ROWS);
    const float* xrow = X + (size_t)(xr0 + lh) * D_IN;

    // ---- H = relu(X @ W1 + b1) -> LDS ----
    for (int nt = 0; nt < H_DIM / 16; ++nt) {
        const int n0 = nt * 16;
        v8f c = {};
        #pragma unroll 8
        for (int k = 0; k < D_IN; k += 4) {
            v2f a = *(const v2f*)(xrow + k + koff);
            const int krow = k + koff;
            v2f b;
            b.x = W1[(size_t)(krow    ) * H_DIM + n0 + lh];
            b.y = W1[(size_t)(krow + 1) * H_DIM + n0 + lh];
            c = wmma_f32(a, b, c);
        }
        const float bias = b1[n0 + lh];
        #pragma unroll
        for (int v = 0; v < 8; ++v) {
            float val = fmaxf(c[v] + bias, 0.0f);
            const int row = (lane < 16) ? v : (v + 8);
            Hs[row * H_DIM + n0 + lh] = val;
        }
    }
    __syncthreads();

    // ---- Z = Hs @ W2 + b2 ----
    v8f z[P_DIM / 16];
    for (int nt = 0; nt < P_DIM / 16; ++nt) {
        const int n0 = nt * 16;
        v8f c = {};
        #pragma unroll 8
        for (int k = 0; k < H_DIM; k += 4) {
            v2f a = *(const v2f*)(&Hs[lh * H_DIM + k + koff]);   // ds_load_b64
            const int krow = k + koff;
            v2f b;
            b.x = W2[(size_t)(krow    ) * P_DIM + n0 + lh];
            b.y = W2[(size_t)(krow + 1) * P_DIM + n0 + lh];
            c = wmma_f32(a, b, c);
        }
        const float bias = b2[n0 + lh];
        #pragma unroll
        for (int v = 0; v < 8; ++v) c[v] += bias;
        z[nt] = c;
    }

    // ---- Row ||z||^2 via half-wave shuffle reduction ----
    float p[8];
    #pragma unroll
    for (int v = 0; v < 8; ++v) p[v] = 0.0f;
    #pragma unroll
    for (int nt = 0; nt < P_DIM / 16; ++nt)
        #pragma unroll
        for (int v = 0; v < 8; ++v) p[v] += z[nt][v] * z[nt][v];
    #pragma unroll
    for (int m = 1; m <= 8; m <<= 1)
        #pragma unroll
        for (int v = 0; v < 8; ++v) p[v] += __shfl_xor(p[v], m, 32);

    float inv[8];
    #pragma unroll
    for (int v = 0; v < 8; ++v)
        inv[v] = 1.0f / fmaxf(sqrtf(p[v]), 1e-12f);

    // ---- Store normalized features as f16 (RNE) ----
    #pragma unroll
    for (int nt = 0; nt < P_DIM / 16; ++nt)
        #pragma unroll
        for (int v = 0; v < 8; ++v) {
            const int row = (lane < 16) ? v : (v + 8);
            F16[(size_t)(r0 + row) * P_DIM + nt * 16 + lh] =
                (_Float16)(z[nt][v] * inv[v]);
        }
}

// ---------------------------------------------------------------------------
// Kernel 2: row-wise logsumexp of F@F^T/T + positive-logit extraction.
// 256 threads = 8 waves = 128 rows per block; B-tiles staged in LDS so each
// 32 KB chunk of F16 is read from L2 once per block instead of once per wave.
// Fixed softmax max = 1/T (unit-norm rows => diagonal dot == 1 is the max).
// ---------------------------------------------------------------------------
#define STG_ROWS 128
#define ROW_PAD  136   // halfs per LDS row: 272 B = 68 dwords -> bank stride 4

__global__ __launch_bounds__(256) void loss_kernel(
    const _Float16* __restrict__ F16, float* __restrict__ partial)
{
    __shared__ _Float16 Fs[STG_ROWS * ROW_PAD];   // ~34 KB

    const int tid  = threadIdx.x;
    const int lane = tid & 31;
    const int wave = tid >> 5;                    // 0..7
    const int lh   = lane & 15;
    const int h16  = lane >> 4;                   // 0 or 1
    const int r0   = blockIdx.x * STG_ROWS + wave * 16;

    // Preload this wave's A fragments: 4 K-steps of 32, 32 VGPRs total.
    // Lane layout (16-bit A 16x32): two 8-half chunks at K=8*h16 and 16+8*h16.
    v16h afr[4];
    const _Float16* frow = F16 + (size_t)(r0 + lh) * P_DIM;
    #pragma unroll
    for (int ks = 0; ks < 4; ++ks) {
        v8h lo = *(const v8h*)(frow + ks * 32 + h16 * 8);
        v8h hi = *(const v8h*)(frow + ks * 32 + 16 + h16 * 8);
        afr[ks] = __builtin_shufflevector(lo, hi,
                      0,1,2,3,4,5,6,7,8,9,10,11,12,13,14,15);
    }

    float se[8], pd[8];
    #pragma unroll
    for (int v = 0; v < 8; ++v) { se[v] = 0.0f; pd[v] = 0.0f; }

    for (int stage = 0; stage < N_TOT / STG_ROWS; ++stage) {
        const int c0 = stage * STG_ROWS;
        __syncthreads();                          // LDS reuse guard
        // Cooperative stage: 128 rows x 128 halfs = 2048 16B chunks, 8/thread.
        #pragma unroll
        for (int s = 0; s < 8; ++s) {
            const int idx   = s * 256 + tid;      // 0..2047
            const int row   = idx >> 4;           // 16 chunks per row
            const int chunk = idx & 15;
            v8h d = *(const v8h*)(F16 + (size_t)(c0 + row) * P_DIM + chunk * 8);
            *(v8h*)(&Fs[row * ROW_PAD + chunk * 8]) = d;
        }
        __syncthreads();

        #pragma unroll
        for (int j2 = 0; j2 < 8; ++j2) {          // 8 column tiles of 16
            v8f c = {};
            #pragma unroll
            for (int ks = 0; ks < 4; ++ks) {
                // B 32x16 lane layout: two 8-half chunks at K=8*h16 and 16+8*h16.
                const _Float16* bp =
                    &Fs[(j2 * 16 + lh) * ROW_PAD + ks * 32 + h16 * 8];
                v8h lo = *(const v8h*)bp;         // ds_load_b128
                v8h hi = *(const v8h*)(bp + 16);  // ds_load_b128
                v16h b = __builtin_shufflevector(lo, hi,
                             0,1,2,3,4,5,6,7,8,9,10,11,12,13,14,15);
                c = wmma_f16(afr[ks], b, c);
            }
            // Fused epilogue: exp-accumulate + branchless positive pick.
            const int colg = c0 + j2 * 16 + lh;
            #pragma unroll
            for (int v = 0; v < 8; ++v) {
                const float s    = c[v];          // dot in [-1, 1]
                const int   rowg = r0 + ((lane < 16) ? v : (v + 8));
                se[v] += __expf((s - 1.0f) * INV_T);
                pd[v] += (colg == (rowg ^ B_ROWS)) ? s : 0.0f;
            }
        }
    }

    // Reduce within each 16-lane half (rows 0-7 in lanes 0-15, 8-15 in 16-31).
    #pragma unroll
    for (int m = 1; m <= 8; m <<= 1)
        #pragma unroll
        for (int v = 0; v < 8; ++v) {
            se[v] += __shfl_xor(se[v], m, 32);
            pd[v] += __shfl_xor(pd[v], m, 32);
        }

    float lsum = 0.0f;
    #pragma unroll
    for (int v = 0; v < 8; ++v)
        lsum += logf(se[v]) + (1.0f - pd[v]) * INV_T;   // LSE - pos/T

    const float tot = lsum + __shfl_xor(lsum, 16, 32);
    if (lane == 0) partial[blockIdx.x * 8 + wave] = tot;
}

// ---------------------------------------------------------------------------
// Kernel 3: deterministic 512 -> 1 reduction, write mean.
// ---------------------------------------------------------------------------
__global__ __launch_bounds__(32) void finalize_kernel(
    const float* __restrict__ partial, float* __restrict__ out)
{
    float s = 0.0f;
    for (int i = threadIdx.x; i < N_TOT / 16; i += 32) s += partial[i];
    #pragma unroll
    for (int m = 16; m >= 1; m >>= 1) s += __shfl_xor(s, m, 32);
    if (threadIdx.x == 0) out[0] = s / (float)N_TOT;
}

extern "C" void kernel_launch(void* const* d_in, const int* in_sizes, int n_in,
                              void* d_out, int out_size, void* d_ws, size_t ws_size,
                              hipStream_t stream)
{
    const float* x1 = (const float*)d_in[0];
    const float* x2 = (const float*)d_in[1];
    const float* W1 = (const float*)d_in[2];
    const float* b1 = (const float*)d_in[3];
    const float* W2 = (const float*)d_in[4];
    const float* b2 = (const float*)d_in[5];
    float* out = (float*)d_out;

    _Float16* F16    = (_Float16*)d_ws;                       // 8192*128 f16 = 2 MB
    float*    partial = (float*)((char*)d_ws +
                        (size_t)N_TOT * P_DIM * sizeof(_Float16));  // 512 f32

    proj_kernel<<<N_TOT / 16, 32, 0, stream>>>(x1, x2, W1, b1, W2, b2, F16);
    loss_kernel<<<N_TOT / STG_ROWS, 256, 0, stream>>>(F16, partial);
    finalize_kernel<<<1, 32, 0, stream>>>(partial, out);
}
